// DGMGearnet_edge_17033840295951
// MI455X (gfx1250) — matrix operand order repaired and checked by reading, not compiled, not run on hardware
//
#include <hip/hip_runtime.h>

#define NN 2048
#define RR 7
#define AR 5
#define SPR 2
#define HH 4
#define DD 512
#define DHH 128
#define TOPK 32
#define WIN 128
#define NEDGE 65536
#define WCOLS 272   // banded score storage: 17 tiles of 16 around each row tile

typedef __attribute__((ext_vector_type(16))) __bf16 v16bf;
typedef __attribute__((ext_vector_type(2)))  __bf16 v2bf;
typedef __attribute__((ext_vector_type(8)))  float  v8f;

union BfFrag { unsigned u[8]; v16bf v; };

// ---------- scalar helpers ----------
__device__ __forceinline__ unsigned pack2bf(float a, float b) {
  v2bf t = { (__bf16)a, (__bf16)b };        // -> v_cvt_pk_bf16_f32
  return __builtin_bit_cast(unsigned, t);
}

__device__ __forceinline__ float wave_fmax(float v) {
#pragma unroll
  for (int off = 16; off > 0; off >>= 1) v = fmaxf(v, __shfl_xor(v, off, 32));
  return v;
}
__device__ __forceinline__ float wave_fsum(float v) {
#pragma unroll
  for (int off = 16; off > 0; off >>= 1) v += __shfl_xor(v, off, 32);
  return v;
}

// ---------- WMMA fragment layout (wave32) ----------
// A (16x32 bf16): half=lane>>4, row M=lane&15.
//   dwords [k0/2 + half*4 .. +3]  = VGPR0..3, dwords [k0/2+8+half*4 .. +3] = VGPR4..7
//   -> two contiguous uint4 loads per K-step.
// B (32x16 bf16): col N=lane&15, dwords [k0/2 + half*4 .. +3], [k0/2+8+half*4 .. +3] -> two uint4 loads.
// C/D: VGPR v -> row M = v + 8*half, col N = lane&15.

template <int NT>
__device__ __forceinline__ void wmma_tiles(const uint4* A4, const uint4* B4, int ldb,
                                           int kdim, int half, v8f* acc) {
  const int jstride = 2 * ldb;  // uint4 stride between adjacent 16-col tiles (16*ldb/8)
  for (int k0 = 0; k0 < kdim; k0 += 32) {
    BfFrag a;
    uint4 alo = A4[(k0 >> 3) + half];
    uint4 ahi = A4[(k0 >> 3) + 2 + half];
    a.u[0] = alo.x; a.u[1] = alo.y; a.u[2] = alo.z; a.u[3] = alo.w;
    a.u[4] = ahi.x; a.u[5] = ahi.y; a.u[6] = ahi.z; a.u[7] = ahi.w;
#pragma unroll
    for (int j = 0; j < NT; ++j) {
      BfFrag b;
      uint4 blo = B4[(k0 >> 3) + half + j * jstride];
      uint4 bhi = B4[(k0 >> 3) + 2 + half + j * jstride];
      b.u[0] = blo.x; b.u[1] = blo.y; b.u[2] = blo.z; b.u[3] = blo.w;
      b.u[4] = bhi.x; b.u[5] = bhi.y; b.u[6] = bhi.z; b.u[7] = bhi.w;
      acc[j] = __builtin_amdgcn_wmma_f32_16x16x32_bf16(false, a.v, false, b.v,
                                                       (short)0, acc[j], false, false);
    }
  }
}

template <int NT>
__device__ __forceinline__ void wmma_tiles_f32A(const float4* A4, const uint4* B4, int ldb,
                                                int kdim, int half, v8f* acc) {
  const int jstride = 2 * ldb;
  for (int k0 = 0; k0 < kdim; k0 += 32) {
    float4 f0 = A4[(k0 >> 2) + half * 2];
    float4 f1 = A4[(k0 >> 2) + 1 + half * 2];
    float4 f2 = A4[(k0 >> 2) + 4 + half * 2];
    float4 f3 = A4[(k0 >> 2) + 5 + half * 2];
    BfFrag a;
    a.u[0] = pack2bf(f0.x, f0.y); a.u[1] = pack2bf(f0.z, f0.w);
    a.u[2] = pack2bf(f1.x, f1.y); a.u[3] = pack2bf(f1.z, f1.w);
    a.u[4] = pack2bf(f2.x, f2.y); a.u[5] = pack2bf(f2.z, f2.w);
    a.u[6] = pack2bf(f3.x, f3.y); a.u[7] = pack2bf(f3.z, f3.w);
#pragma unroll
    for (int j = 0; j < NT; ++j) {
      BfFrag b;
      uint4 blo = B4[(k0 >> 3) + half + j * jstride];
      uint4 bhi = B4[(k0 >> 3) + 2 + half + j * jstride];
      b.u[0] = blo.x; b.u[1] = blo.y; b.u[2] = blo.z; b.u[3] = blo.w;
      b.u[4] = bhi.x; b.u[5] = bhi.y; b.u[6] = bhi.z; b.u[7] = bhi.w;
      acc[j] = __builtin_amdgcn_wmma_f32_16x16x32_bf16(false, a.v, false, b.v,
                                                       (short)0, acc[j], false, false);
    }
  }
}

__device__ __forceinline__ void store_tile_f32(float* C, int ldc, int m0, int n0,
                                               int lane, v8f acc, bool relu) {
  int half = lane >> 4, l15 = lane & 15;
#pragma unroll
  for (int v = 0; v < 8; ++v) {
    float f = acc[v];
    if (relu) f = fmaxf(f, 0.f);
    C[(size_t)(m0 + v + 8 * half) * ldc + n0 + l15] = f;
  }
}
__device__ __forceinline__ void store_tile_bf16(__bf16* C, int ldc, int m0, int n0,
                                                int lane, v8f acc, bool relu) {
  int half = lane >> 4, l15 = lane & 15;
#pragma unroll
  for (int v = 0; v < 8; ++v) {
    float f = acc[v];
    if (relu) f = fmaxf(f, 0.f);
    C[(size_t)(m0 + v + 8 * half) * ldc + n0 + l15] = (__bf16)f;
  }
}

// ---------- prep kernels ----------
__global__ void k_scatter(const int* ni, const int* no, const int* rel,
                          const float* w, float* adjF) {
  int e = blockIdx.x * blockDim.x + threadIdx.x;
  if (e < NEDGE) {
    size_t idx = (size_t)ni[e] * (RR * NN) + (size_t)rel[e] * NN + (size_t)no[e];
    atomicAdd(adjF + idx, w[e]);
  }
}

__global__ void k_prep_x(const float* x, __bf16* xT, __bf16* gmsgX) {
  int t = blockIdx.x * blockDim.x + threadIdx.x;
  if (t < NN * DD) {
    int j = t / DD, d = t % DD;
    __bf16 b = (__bf16)x[t];
    xT[(size_t)d * NN + j] = b;                                   // x^T  (D x N)
    gmsgX[(size_t)j * ((RR + 1) * DD) + RR * DD + d] = b;         // slot R holds x
  }
}

__global__ void k_transpose_w(const float* W, __bf16* Wt, int nmat) {
  int t = blockIdx.x * blockDim.x + threadIdx.x;
  if (t < nmat * DD * DD) {
    int mat = t / (DD * DD), r2 = t % (DD * DD);
    int e = r2 / DD, d = r2 % DD;
    Wt[t] = (__bf16)W[(size_t)mat * DD * DD + (size_t)d * DD + e]; // Wt[mat][e][d]
  }
}

__global__ void k_prep_wgs(const float* Wgear, const float* Wself, __bf16* WgsT) {
  int t = blockIdx.x * blockDim.x + threadIdx.x;
  if (t < DD * (RR + 1) * DD) {
    int e = t / ((RR + 1) * DD), c = t % ((RR + 1) * DD);
    int s = c / DD, d = c % DD;
    float v = (s < RR) ? Wgear[(size_t)s * DD * DD + (size_t)d * DD + e]
                       : Wself[(size_t)d * DD + e];
    WgsT[t] = (__bf16)v;                                          // WgsT[e][s*D+d]
  }
}

// ---------- GEMM kernels (one wave = 16x64 output strip, 4 accumulators) ----------
__global__ void k_msg_gemm(const float* adjF, const __bf16* xT, __bf16* msgB) {
  int w = blockIdx.x * (blockDim.x >> 5) + (threadIdx.x >> 5);
  int lane = threadIdx.x & 31;
  int tn4 = w & 7; w >>= 3; int tm = w & 127; int ar = w >> 7;
  if (ar >= AR) return;
  int half = lane >> 4, l15 = lane & 15;
  int m0 = tm * 16, n0 = tn4 * 64;
  const float4* A4 = (const float4*)(adjF + (size_t)(m0 + l15) * (RR * NN)
                                     + (size_t)(ar + SPR) * NN);
  const uint4* B4 = (const uint4*)(xT + (size_t)(n0 + l15) * NN);
  v8f acc[4] = {};
  wmma_tiles_f32A<4>(A4, B4, NN, NN, half, acc);
#pragma unroll
  for (int j = 0; j < 4; ++j)
    store_tile_bf16(msgB + (size_t)ar * NN * DD, DD, m0, n0 + 16 * j, lane, acc[j], false);
}

__global__ void k_relout_gemm(const __bf16* msgB, const __bf16* WrelT, __bf16* relB) {
  int w = blockIdx.x * (blockDim.x >> 5) + (threadIdx.x >> 5);
  int lane = threadIdx.x & 31;
  int tn4 = w & 7; w >>= 3; int tm = w & 127; int ar = w >> 7;
  if (ar >= AR) return;
  int half = lane >> 4, l15 = lane & 15;
  int m0 = tm * 16, n0 = tn4 * 64;
  const uint4* A4 = (const uint4*)(msgB + ((size_t)ar * NN + m0 + l15) * DD);
  const uint4* B4 = (const uint4*)(WrelT + (size_t)(ar + SPR) * DD * DD
                                   + (size_t)(n0 + l15) * DD);
  v8f acc[4] = {};
  wmma_tiles<4>(A4, B4, DD, DD, half, acc);
#pragma unroll
  for (int j = 0; j < 4; ++j)
    store_tile_bf16(relB + (size_t)ar * NN * DD, DD, m0, n0 + 16 * j, lane, acc[j], true);
}

__global__ void k_qk_gemm(const __bf16* relB, const __bf16* WqT, const __bf16* WkT,
                          __bf16* Qb, __bf16* Kb) {
  int w = blockIdx.x * (blockDim.x >> 5) + (threadIdx.x >> 5);
  int lane = threadIdx.x & 31;
  int tn4 = w & 7; w >>= 3; int tm = w & 127; w >>= 7;
  int ar = w % AR, z = w / AR;
  if (z >= 2) return;
  int half = lane >> 4, l15 = lane & 15;
  int m0 = tm * 16, n0 = tn4 * 64;
  const __bf16* Wt = z ? WkT : WqT;
  __bf16* Out = z ? Kb : Qb;
  const uint4* A4 = (const uint4*)(relB + ((size_t)ar * NN + m0 + l15) * DD);
  const uint4* B4 = (const uint4*)(Wt + (size_t)(n0 + l15) * DD);
  v8f acc[4] = {};
  wmma_tiles<4>(A4, B4, DD, DD, half, acc);
#pragma unroll
  for (int j = 0; j < 4; ++j)
    store_tile_bf16(Out + (size_t)ar * NN * DD, DD, m0, n0 + 16 * j, lane, acc[j], false);
}

__global__ void k_scores(const __bf16* Qb, const __bf16* Kb, float* Sw) {
  int w = blockIdx.x * (blockDim.x >> 5) + (threadIdx.x >> 5);
  int lane = threadIdx.x & 31;
  int tmt = w % 17; w /= 17;
  int tm = w & 127; w >>= 7;
  int h = w & 3; int ar = w >> 2;
  if (ar >= AR) return;
  int half = lane >> 4, l15 = lane & 15;
  int n0 = tm * 16;
  int m0 = n0 - 128 + tmt * 16;
  int mcol = m0 + l15;
  int mc = mcol < 0 ? 0 : (mcol >= NN ? NN - 1 : mcol);
  const uint4* A4 = (const uint4*)(Qb + ((size_t)ar * NN + n0 + l15) * DD + h * DHH);
  const uint4* B4 = (const uint4*)(Kb + ((size_t)ar * NN + mc) * DD + h * DHH);
  v8f acc[1] = {};
  wmma_tiles<1>(A4, B4, DD, DHH, half, acc);
#pragma unroll
  for (int v = 0; v < 8; ++v) {
    int nn = n0 + v + 8 * half;
    float s = acc[0][v] * 0.17677669529663687f;   // 1/(sqrt(128)*0.5)
    int dd = nn - mcol; if (dd < 0) dd = -dd;
    if (mcol < 0 || mcol >= NN || dd > WIN) s = -1.0e9f;
    Sw[((size_t)(ar * HH + h) * NN + nn) * WCOLS + (tmt * 16 + l15)] = s;
  }
}

// ---------- softmax + top-32 threshold + head mean (one wave per (ar,n) row) ----------
__global__ void k_softmax_topk(const float* Sw, float* Pm) {
  int w = blockIdx.x * (blockDim.x >> 5) + (threadIdx.x >> 5);
  int lane = threadIdx.x & 31;
  if (w >= AR * NN) return;
  int ar = w / NN, n = w % NN;
  float acc[9];
#pragma unroll
  for (int i = 0; i < 9; ++i) acc[i] = 0.f;
  for (int h = 0; h < HH; ++h) {
    const float* Srow = Sw + ((size_t)(ar * HH + h) * NN + n) * WCOLS;
    float pv[9];
    float mx = -3.0e38f;
#pragma unroll
    for (int i = 0; i < 9; ++i) {
      int idx = lane + 32 * i;
      float v = (idx < WCOLS) ? Srow[idx] : -3.0e38f;
      pv[i] = v;
      mx = fmaxf(mx, v);
    }
    mx = wave_fmax(mx);
    float ssum = 0.f;
#pragma unroll
    for (int i = 0; i < 9; ++i) {
      int idx = lane + 32 * i;
      float e = (idx < WCOLS) ? __expf(pv[i] - mx) : 0.f;
      pv[i] = e; ssum += e;
    }
    ssum = wave_fsum(ssum);
    float inv = 1.f / ssum;
#pragma unroll
    for (int i = 0; i < 9; ++i) pv[i] *= inv;
    // iterative extraction of the 32 largest; kth = value of the 32nd
    float wk[9];
#pragma unroll
    for (int i = 0; i < 9; ++i) { int idx = lane + 32 * i; wk[i] = (idx < WCOLS) ? pv[i] : -1.f; }
    float kth = 0.f;
    for (int it = 0; it < TOPK; ++it) {
      float lm = -1.f; int ls = 0;
#pragma unroll
      for (int i = 0; i < 9; ++i) if (wk[i] > lm) { lm = wk[i]; ls = i; }
      float gm = wave_fmax(lm);
      kth = gm;
      unsigned long long ball = __ballot(lm == gm);
      int src = (int)__ffsll(ball) - 1;
      if (lane == src) wk[ls] = -1.f;
    }
#pragma unroll
    for (int i = 0; i < 9; ++i) {
      int idx = lane + 32 * i;
      if (idx < WCOLS && pv[i] >= kth) acc[i] += pv[i] * 0.25f;  // mean over H=4 heads
    }
  }
  float* Prow = Pm + (size_t)w * WCOLS;
#pragma unroll
  for (int i = 0; i < 9; ++i) { int idx = lane + 32 * i; if (idx < WCOLS) Prow[idx] = acc[i]; }
}

// ---------- new adjacency: relation permutation + max with attention ----------
__global__ void k_newadj(const float* adjF, const float* Pm, __bf16* newA) {
  size_t t = (size_t)blockIdx.x * blockDim.x + threadIdx.x;
  if (t >= (size_t)NN * RR * NN) return;
  int m = (int)(t % NN); size_t t2 = t / NN;
  int r = (int)(t2 % RR); int n = (int)(t2 / RR);
  int sr = (r < AR) ? (r + SPR) : (r - AR);
  float v = adjF[((size_t)n * RR + sr) * NN + m];
  if (r < AR) {
    int dd = n - m; if (dd < 0) dd = -dd;
    if (dd <= WIN) {
      int nt0 = n & ~15;
      v = fmaxf(v, Pm[((size_t)r * NN + n) * WCOLS + (m - (nt0 - 128))]);
    }
  }
  newA[t] = (__bf16)v;
}

__global__ void k_gmsg_gemm(const __bf16* newA, const __bf16* xT, __bf16* gmsgX) {
  int w = blockIdx.x * (blockDim.x >> 5) + (threadIdx.x >> 5);
  int lane = threadIdx.x & 31;
  int tn4 = w & 7; w >>= 3; int tm = w & 127; int s = w >> 7;
  if (s >= RR) return;
  int half = lane >> 4, l15 = lane & 15;
  int m0 = tm * 16, n0 = tn4 * 64;
  const uint4* A4 = (const uint4*)(newA + ((size_t)(m0 + l15) * RR + s) * NN);
  const uint4* B4 = (const uint4*)(xT + (size_t)(n0 + l15) * NN);
  v8f acc[4] = {};
  wmma_tiles<4>(A4, B4, NN, NN, half, acc);
#pragma unroll
  for (int j = 0; j < 4; ++j)
    store_tile_bf16(gmsgX + (size_t)s * DD, (RR + 1) * DD, m0, n0 + 16 * j, lane, acc[j], false);
}

__global__ void k_hidden_gemm(const __bf16* gmsgX, const __bf16* WgsT, float* nodeF) {
  int w = blockIdx.x * (blockDim.x >> 5) + (threadIdx.x >> 5);
  int lane = threadIdx.x & 31;
  int tn4 = w & 7; int tm = w >> 3;
  if (tm >= 128) return;
  int half = lane >> 4, l15 = lane & 15;
  int m0 = tm * 16, n0 = tn4 * 64;
  const int KD = (RR + 1) * DD;  // 4096
  const uint4* A4 = (const uint4*)(gmsgX + (size_t)(m0 + l15) * KD);
  const uint4* B4 = (const uint4*)(WgsT + (size_t)(n0 + l15) * KD);
  v8f acc[4] = {};
  wmma_tiles<4>(A4, B4, KD, KD, half, acc);
#pragma unroll
  for (int j = 0; j < 4; ++j)
    store_tile_f32(nodeF, DD, m0, n0 + 16 * j, lane, acc[j], true);  // relu
}

__global__ void k_graphsum(const float* nodeF, float* graphF) {
  int c = blockIdx.x * blockDim.x + threadIdx.x;
  if (c < DD) {
    float s = 0.f;
    for (int i = 0; i < NN; ++i) s += nodeF[(size_t)i * DD + c];
    graphF[c] = s;
  }
}

// ---------- launcher ----------
extern "C" void kernel_launch(void* const* d_in, const int* in_sizes, int n_in,
                              void* d_out, int out_size, void* d_ws, size_t ws_size,
                              hipStream_t stream) {
  (void)in_sizes; (void)n_in; (void)out_size; (void)ws_size;
  const float* x      = (const float*)d_in[0];
  const int* node_in  = (const int*)d_in[1];
  const int* node_out = (const int*)d_in[2];
  const int* rel      = (const int*)d_in[3];
  const float* ew     = (const float*)d_in[4];
  const float* Wrel   = (const float*)d_in[5];
  const float* Wq     = (const float*)d_in[6];
  const float* Wk     = (const float*)d_in[7];
  const float* Wgear  = (const float*)d_in[8];
  const float* Wself  = (const float*)d_in[9];

  char* p = (char*)d_ws;
  auto alloc = [&](size_t bytes) -> char* {
    char* q = p; p += (bytes + 255) & ~(size_t)255; return q;
  };
  float*  adjF  = (float*) alloc((size_t)NN * RR * NN * 4);
  __bf16* xT    = (__bf16*)alloc((size_t)DD * NN * 2);
  __bf16* WrelT = (__bf16*)alloc((size_t)RR * DD * DD * 2);
  __bf16* WqT   = (__bf16*)alloc((size_t)DD * DD * 2);
  __bf16* WkT   = (__bf16*)alloc((size_t)DD * DD * 2);
  __bf16* WgsT  = (__bf16*)alloc((size_t)DD * (RR + 1) * DD * 2);
  __bf16* msgB  = (__bf16*)alloc((size_t)AR * NN * DD * 2);
  __bf16* relB  = (__bf16*)alloc((size_t)AR * NN * DD * 2);
  __bf16* Qb    = (__bf16*)alloc((size_t)AR * NN * DD * 2);
  __bf16* Kb    = (__bf16*)alloc((size_t)AR * NN * DD * 2);
  float*  Sw    = (float*) alloc((size_t)AR * HH * NN * WCOLS * 4);
  float*  Pm    = (float*) alloc((size_t)AR * NN * WCOLS * 4);
  __bf16* newA  = (__bf16*)alloc((size_t)NN * RR * NN * 2);
  __bf16* gmsgX = (__bf16*)alloc((size_t)NN * (RR + 1) * DD * 2);

  float* graphF = (float*)d_out;
  float* nodeF  = (float*)d_out + DD;

  hipMemsetAsync(adjF, 0, (size_t)NN * RR * NN * 4, stream);
  k_scatter<<<NEDGE / 256, 256, 0, stream>>>(node_in, node_out, rel, ew, adjF);
  k_prep_x<<<(NN * DD) / 256, 256, 0, stream>>>(x, xT, gmsgX);
  k_transpose_w<<<(RR * DD * DD) / 256, 256, 0, stream>>>(Wrel, WrelT, RR);
  k_transpose_w<<<(DD * DD) / 256, 256, 0, stream>>>(Wq, WqT, 1);
  k_transpose_w<<<(DD * DD) / 256, 256, 0, stream>>>(Wk, WkT, 1);
  k_prep_wgs<<<(DD * (RR + 1) * DD) / 256, 256, 0, stream>>>(Wgear, Wself, WgsT);

  k_msg_gemm    <<<(AR * 128 * 8) / 8, 256, 0, stream>>>(adjF, xT, msgB);
  k_relout_gemm <<<(AR * 128 * 8) / 8, 256, 0, stream>>>(msgB, WrelT, relB);
  k_qk_gemm     <<<(2 * AR * 128 * 8) / 8, 256, 0, stream>>>(relB, WqT, WkT, Qb, Kb);
  k_scores      <<<(AR * HH * 128 * 17) / 8, 256, 0, stream>>>(Qb, Kb, Sw);
  k_softmax_topk<<<(AR * NN) / 8, 256, 0, stream>>>(Sw, Pm);
  k_newadj      <<<(int)(((size_t)NN * RR * NN) / 256), 256, 0, stream>>>(adjF, Pm, newA);
  k_gmsg_gemm   <<<(RR * 128 * 8) / 8, 256, 0, stream>>>(newA, xT, gmsgX);
  k_hidden_gemm <<<(128 * 8) / 8, 256, 0, stream>>>(gmsgX, WgsT, nodeF);
  k_graphsum    <<<(DD + 255) / 256, 256, 0, stream>>>(nodeF, graphF);
}
